// BNNVGG11_28518582846034
// MI455X (gfx1250) — compile-verified, matrix operand order checked by reading, not run on hardware
//
#include <hip/hip_runtime.h>
#include <math.h>

typedef __attribute__((ext_vector_type(8))) int v8i;

#define NB 256
#define EPSF 1e-5f
#define USE_ASYNC 1

static __device__ __forceinline__ float sgnf(float v){ return v >= 0.f ? 1.f : -1.f; }

// ---- CDNA5 async global->LDS staging (ASYNCcnt) with sync fallback ----------
#if USE_ASYNC
static __device__ __forceinline__ void stage16(void* lp, const void* gp) {
    // Low 32 bits of a generic shared pointer are the LDS byte offset.
    unsigned lo = (unsigned)(unsigned long long)(uintptr_t)lp;
    asm volatile("global_load_async_to_lds_b128 %0, %1, off" :: "v"(lo), "v"(gp) : "memory");
}
static __device__ __forceinline__ void stage_wait() {
    asm volatile("s_wait_asynccnt 0x0" ::: "memory");
}
#else
static __device__ __forceinline__ void stage16(void* lp, const void* gp) {
    *(int4*)lp = *(const int4*)gp;
}
static __device__ __forceinline__ void stage_wait() {}
#endif

// LDS B-chunk layout: 64 rows (N) x 64 bytes (K), padded to 80B stride
// -> 16 fragment rows hit distinct banks (20k mod 64 distinct), b128-aligned.
#define BROW 80
#define BBUF (64 * BROW)

// ---------------------------------------------------------------------------
// Layer 0: real-valued input (3ch) x sign(W), fused 2x2 maxpool epilogue.
// ---------------------------------------------------------------------------
__global__ void conv0_pool_kernel(const float* __restrict__ x,    // [256,3,32,32] NCHW
                                  const float* __restrict__ w,    // [64,3,3,3]
                                  const float* __restrict__ bias, // [64]
                                  float* __restrict__ outP)       // [256,16,16,64] NHWC
{
    int idx = blockIdx.x * blockDim.x + threadIdx.x;
    const int total = NB * 16 * 16 * 64;
    if (idx >= total) return;
    int co = idx & 63;
    int t  = idx >> 6;
    int ox = t & 15; t >>= 4;
    int oy = t & 15; t >>= 4;
    int n  = t;
    float wl[27];
#pragma unroll
    for (int q = 0; q < 27; ++q) wl[q] = sgnf(w[co * 27 + q]);
    float best = -INFINITY;
#pragma unroll
    for (int sy = 0; sy < 2; ++sy)
#pragma unroll
    for (int sx = 0; sx < 2; ++sx) {
        int yy = oy * 2 + sy, xx = ox * 2 + sx;
        float acc = bias[co];
        for (int ci = 0; ci < 3; ++ci)
            for (int ky = 0; ky < 3; ++ky) {
                int iy = yy + ky - 1;
                if ((unsigned)iy >= 32u) continue;
                for (int kx = 0; kx < 3; ++kx) {
                    int ix = xx + kx - 1;
                    if ((unsigned)ix >= 32u) continue;
                    acc += x[((n * 3 + ci) * 32 + iy) * 32 + ix] * wl[(ci * 3 + ky) * 3 + kx];
                }
            }
        best = fmaxf(best, acc);
    }
    outP[idx] = best;
}

// Binarize conv weights [Co,Ci,3,3] -> int8 [Co, 9*Ci], K-order (ky,kx,ci)
__global__ void wprep_conv_kernel(const float* __restrict__ w, signed char* __restrict__ wb,
                                  int Co, int Ci)
{
    int K = 9 * Ci;
    long idx = (long)blockIdx.x * blockDim.x + threadIdx.x;
    if (idx >= (long)Co * K) return;
    int k  = (int)(idx % K);
    int co = (int)(idx / K);
    int ci = k % Ci;
    int kk = k / Ci;
    int ky = kk / 3, kx = kk - 3 * ky;
    wb[idx] = (w[((co * Ci + ci) * 3 + ky) * 3 + kx] >= 0.f) ? (signed char)1 : (signed char)-1;
}

// Binarize linear weights [Co,Ci] -> int8 [CoPad,Ci], zero-padded rows.
__global__ void wprep_lin_kernel(const float* __restrict__ w, signed char* __restrict__ wb,
                                 int Co, int Ci, int CoPad)
{
    long idx = (long)blockIdx.x * blockDim.x + threadIdx.x;
    if (idx >= (long)CoPad * Ci) return;
    int ci = (int)(idx % Ci);
    int co = (int)(idx / Ci);
    wb[idx] = (co < Co) ? ((w[(long)co * Ci + ci] >= 0.f) ? (signed char)1 : (signed char)-1)
                        : (signed char)0;
}

// int4 zero fill (halo init)
__global__ void fill0_kernel(int4* __restrict__ p, long n16)
{
    long idx = (long)blockIdx.x * blockDim.x + threadIdx.x;
    if (idx < n16) { int4 z = {0, 0, 0, 0}; p[idx] = z; }
}

// Load one 16x64 IU8 fragment (4x 8B chunks)
static __device__ __forceinline__ v8i frag_g(const signed char* p)
{
    v8i f;
#pragma unroll
    for (int i = 0; i < 4; ++i) {
        int2 c = *(const int2*)(p + (i << 4));
        f[2 * i] = c.x; f[2 * i + 1] = c.y;
    }
    return f;
}

// ---------------------------------------------------------------------------
// Binary conv, implicit-GEMM on halo-padded int8 NHWC input.
// Block: 8 waves x (32M x 64N) = 256 M-rows x 64 N-cols. B chunk (64x64)
// double-buffered in LDS via async global->LDS, shared by 8 waves; each B
// fragment read from LDS feeds TWO WMMAs (two A fragments, 8 accumulators).
// Buffer phase is pure address arithmetic (bufoff) so the loop body is
// iteration-invariant: no unroll specialization, no accumulator copies.
// ---------------------------------------------------------------------------
__global__ __launch_bounds__(256)
void bconv_iu8_kernel(const signed char* __restrict__ Xpad, // [Nb,H+2,W+2,Ci]
                      const signed char* __restrict__ Bw,   // [Co, 9*Ci]
                      const float* __restrict__ bias,
                      float* __restrict__ Cout_,            // [Nb*H*W, Co]
                      int Nb, int H, int W, int Ci, int Co, int cs, int nkcs)
{                                                           // cs=log2 Ci, nkcs=log2(Ci/64)
    __shared__ signed char Bs[2][BBUF];
    const int K  = 9 * Ci;
    const int Hp = H + 2, Wp = W + 2;
    const int ntn = Co >> 6;
    const int mb = blockIdx.x / ntn, nt = blockIdx.x % ntn;
    const int tid = threadIdx.x;
    const int lane = tid & 31, wave = tid >> 5;
    const int n0 = nt << 6;
    const int m0 = (mb << 8) + (wave << 5);
    const int r16 = lane & 15;
    const int kb  = (lane >> 4) << 3;          // 8-bit A layout: lanes 16-31 hold K+8
    // Two M-rows per lane (tiles m0..m0+15 and m0+16..m0+31)
    const int row0 = m0 + r16;
    const int row1 = row0 + 16;
    int t0 = row0 / W; const int ox0 = row0 - t0 * W; const int oy0 = t0 % H; const int ni0 = t0 / H;
    int t1 = row1 / W; const int ox1 = row1 - t1 * W; const int oy1 = t1 % H; const int ni1 = t1 / H;

    // Cooperative staging coords: thread -> (row 0..63, 16B column 0..3)
    const int sr = tid >> 2;
    const int sc = (tid & 3) << 4;
    const signed char* gstage = Bw + (long)(n0 + sr) * K + sc;
    signed char* const lbase = &Bs[0][sr * BROW + sc];
    const signed char* const bbase = &Bs[0][0];

    stage16(lbase, gstage);
    stage_wait();
    __syncthreads();

    v8i z = {0, 0, 0, 0, 0, 0, 0, 0};
    v8i acc[8];
#pragma unroll
    for (int j = 0; j < 8; ++j) acc[j] = z;

    const int nkc = 1 << nkcs;                 // Ci/64 chunks per tap
    int bufoff = 0;                            // 0 or BBUF; flips each K-step
    int ktg = 0;
#pragma clang loop unroll(disable)
    for (int kk = 0; kk < 9; ++kk) {
        const int ky = kk / 3, kx = kk - 3 * ky;
        const signed char* ap0 =
            Xpad + ((((long)ni0 * Hp + (oy0 + ky)) * Wp + (ox0 + kx)) << cs) + kb;
        const signed char* ap1 =
            Xpad + ((((long)ni1 * Hp + (oy1 + ky)) * Wp + (ox1 + kx)) << cs) + kb;
#pragma clang loop unroll(disable)
        for (int cc = 0; cc < nkc; ++cc) {
            if (ktg + 64 < K)
                stage16(lbase + (BBUF - bufoff), gstage + (long)ktg + 64);

            const int co64 = cc << 6;
            v8i afr0 = frag_g(ap0 + co64);
            v8i afr1 = frag_g(ap1 + co64);

            const signed char* bsb = bbase + bufoff;
#pragma unroll
            for (int j = 0; j < 4; ++j) {
                const signed char* bp = bsb + ((j << 4) + r16) * BROW + kb;
                v8i bfr;
#pragma unroll
                for (int i = 0; i < 4; ++i) {
                    int2 c = *(const int2*)(bp + (i << 4));
                    bfr[2 * i] = c.x; bfr[2 * i + 1] = c.y;
                }
                acc[j]     = __builtin_amdgcn_wmma_i32_16x16x64_iu8(true, afr0, true, bfr,
                                                                    acc[j], false, false);
                acc[j + 4] = __builtin_amdgcn_wmma_i32_16x16x64_iu8(true, afr1, true, bfr,
                                                                    acc[j + 4], false, false);
            }
            stage_wait();
            __syncthreads();
            bufoff = BBUF - bufoff;
            ktg += 64;
        }
    }
    // D layout: VGPR r, lane L -> (M = r + 8*(L>>4), N = L&15)
    const int mrow0 = m0 + ((lane >> 4) ? 8 : 0);
#pragma unroll
    for (int j = 0; j < 4; ++j) {
        int nn = n0 + (j << 4) + r16;
        float bv = bias[nn];
#pragma unroll
        for (int r = 0; r < 8; ++r) {
            Cout_[(long)(mrow0 + r) * Co + nn]      = (float)acc[j][r] + bv;
            Cout_[(long)(mrow0 + 16 + r) * Co + nn] = (float)acc[j + 4][r] + bv;
        }
    }
}

// ---------------------------------------------------------------------------
// Generic binary GEMM (linear layers), same staging + 32x64 wave tile.
// C[M,N] = A[M,K] . B[Npad,K]^T + bias ; store guarded by true N.
// ---------------------------------------------------------------------------
__global__ __launch_bounds__(256)
void bgemm_iu8_kernel(const signed char* __restrict__ A,
                      const signed char* __restrict__ B,
                      const float* __restrict__ bias,
                      float* __restrict__ C, int M, int N, int K)
{
    __shared__ signed char Bs[2][BBUF];
    const int ntn = (N + 63) >> 6;
    const int mb = blockIdx.x / ntn, nt = blockIdx.x % ntn;
    const int tid = threadIdx.x;
    const int lane = tid & 31, wave = tid >> 5;
    const int n0 = nt << 6;
    const int m0 = (mb << 8) + (wave << 5);
    const int r16 = lane & 15;
    const int kb  = (lane >> 4) << 3;
    const signed char* a0 = A + (long)(m0 + r16) * K + kb;
    const signed char* a1 = A + (long)(m0 + 16 + r16) * K + kb;

    const int sr = tid >> 2;
    const int sc = (tid & 3) << 4;
    const signed char* gstage = B + (long)(n0 + sr) * K + sc;
    signed char* const lbase = &Bs[0][sr * BROW + sc];
    const signed char* const bbase = &Bs[0][0];

    stage16(lbase, gstage);
    stage_wait();
    __syncthreads();

    v8i z = {0, 0, 0, 0, 0, 0, 0, 0};
    v8i acc[8];
#pragma unroll
    for (int j = 0; j < 8; ++j) acc[j] = z;

    const int nk = K >> 6;
    int bufoff = 0;
#pragma clang loop unroll(disable)
    for (int step = 0; step < nk; ++step) {
        if (step + 1 < nk)
            stage16(lbase + (BBUF - bufoff), gstage + (long)(step + 1) * 64);

        const int kt = step << 6;
        v8i afr0 = frag_g(a0 + kt);
        v8i afr1 = frag_g(a1 + kt);

        const signed char* bsb = bbase + bufoff;
#pragma unroll
        for (int j = 0; j < 4; ++j) {
            const signed char* bp = bsb + ((j << 4) + r16) * BROW + kb;
            v8i bfr;
#pragma unroll
            for (int i = 0; i < 4; ++i) {
                int2 c = *(const int2*)(bp + (i << 4));
                bfr[2 * i] = c.x; bfr[2 * i + 1] = c.y;
            }
            acc[j]     = __builtin_amdgcn_wmma_i32_16x16x64_iu8(true, afr0, true, bfr,
                                                                acc[j], false, false);
            acc[j + 4] = __builtin_amdgcn_wmma_i32_16x16x64_iu8(true, afr1, true, bfr,
                                                                acc[j + 4], false, false);
        }
        stage_wait();
        __syncthreads();
        bufoff = BBUF - bufoff;
    }
    const int mrow0 = m0 + ((lane >> 4) ? 8 : 0);
#pragma unroll
    for (int j = 0; j < 4; ++j) {
        int nn = n0 + (j << 4) + r16;
        if (nn >= N) continue;
        float bv = bias[nn];
#pragma unroll
        for (int r = 0; r < 8; ++r) {
            C[(long)(mrow0 + r) * N + nn]      = (float)acc[j][r] + bv;
            C[(long)(mrow0 + 16 + r) * N + nn] = (float)acc[j + 4][r] + bv;
        }
    }
}

// 2x2 maxpool on NHWC float
__global__ void pool_kernel(const float* __restrict__ Y, float* __restrict__ P,
                            int Nb, int H, int W, int C)
{
    int Ho = H >> 1, Wo = W >> 1;
    long total = (long)Nb * Ho * Wo * C;
    long idx = (long)blockIdx.x * blockDim.x + threadIdx.x;
    if (idx >= total) return;
    int c  = (int)(idx % C);
    long t = idx / C;
    int ox = (int)(t % Wo); t /= Wo;
    int oy = (int)(t % Ho);
    int n  = (int)(t / Ho);
    const float* base = Y + (((long)n * H + 2 * oy) * W + 2 * ox) * C + c;
    float a = base[0], b = base[(long)C], d = base[(long)W * C], e = base[(long)W * C + C];
    P[idx] = fmaxf(fmaxf(a, b), fmaxf(d, e));
}

// Per-channel batch stats (training-mode BN): mean[c], inv[c]=rsqrt(var+eps)
__global__ void stats_kernel(const float* __restrict__ X, float* __restrict__ mean,
                             float* __restrict__ inv, int Mrows, int C)
{
    int c = blockIdx.x;
    float s = 0.f, ss = 0.f;
    for (int m = threadIdx.x; m < Mrows; m += blockDim.x) {
        float v = X[(long)m * C + c];
        s += v; ss += v * v;
    }
    __shared__ float sh[256], sh2[256];
    sh[threadIdx.x] = s; sh2[threadIdx.x] = ss;
    __syncthreads();
    for (int o = 128; o > 0; o >>= 1) {
        if (threadIdx.x < o) { sh[threadIdx.x] += sh[threadIdx.x + o];
                               sh2[threadIdx.x] += sh2[threadIdx.x + o]; }
        __syncthreads();
    }
    if (threadIdx.x == 0) {
        float mu = sh[0] / (float)Mrows;
        float vr = sh2[0] / (float)Mrows - mu * mu;
        mean[c] = mu;
        inv[c]  = rsqrtf(vr + EPSF);
    }
}

// sign(hardtanh(bn(x))) == sign(bn(x)); optional halo-offset write (pad=1)
__global__ void bnsign_kernel(const float* __restrict__ X, const float* __restrict__ g,
                              const float* __restrict__ b, const float* __restrict__ mean,
                              const float* __restrict__ inv, signed char* __restrict__ out,
                              int Nb, int H, int W, int C, int pad)
{
    long total = (long)Nb * H * W * C;
    long idx = (long)blockIdx.x * blockDim.x + threadIdx.x;
    if (idx >= total) return;
    int c  = (int)(idx % C);
    long t = idx / C;
    int xx = (int)(t % W); t /= W;
    int yy = (int)(t % H);
    int n  = (int)(t / H);
    float zv = g[c] * (X[idx] - mean[c]) * inv[c] + b[c];
    signed char sv = (zv >= 0.f) ? (signed char)1 : (signed char)-1;
    long o = pad ? ((((long)n * (H + 2) + yy + 1) * (W + 2) + xx + 1) * C + c) : idx;
    out[o] = sv;
}

// BN apply keeping float (final classifier head)
__global__ void bnapply_kernel(const float* __restrict__ X, const float* __restrict__ g,
                               const float* __restrict__ b, const float* __restrict__ mean,
                               const float* __restrict__ inv, float* __restrict__ out,
                               long total, int C)
{
    long idx = (long)blockIdx.x * blockDim.x + threadIdx.x;
    if (idx >= total) return;
    int c = (int)(idx % C);
    out[idx] = g[c] * (X[idx] - mean[c]) * inv[c] + b[c];
}

__global__ void logsoftmax_kernel(const float* __restrict__ Z, float* __restrict__ out,
                                  int rows, int C)
{
    int r = blockIdx.x * blockDim.x + threadIdx.x;
    if (r >= rows) return;
    float mx = -INFINITY;
    for (int c = 0; c < C; ++c) mx = fmaxf(mx, Z[r * C + c]);
    float s = 0.f;
    for (int c = 0; c < C; ++c) s += expf(Z[r * C + c] - mx);
    float ls = logf(s);
    for (int c = 0; c < C; ++c) out[r * C + c] = Z[r * C + c] - mx - ls;
}

// ---------------------------------------------------------------------------
// Orchestration. Workspace (~60.1 MB, fully L2-resident on MI455X):
//   Y float (32 MB) | P pooled float (16 MB) | X int8 halo acts (8 MB)
//   WB int8 weights (4 MB) | stats tail
// ---------------------------------------------------------------------------
extern "C" void kernel_launch(void* const* d_in, const int* in_sizes, int n_in,
                              void* d_out, int out_size, void* d_ws, size_t ws_size,
                              hipStream_t stream)
{
    const float* x = (const float*)d_in[0];
    const float *cw[8], *cb[8], *cg[8], *ct[8];
    for (int i = 0; i < 8; ++i) {
        cw[i] = (const float*)d_in[1 + 4 * i];
        cb[i] = (const float*)d_in[2 + 4 * i];
        cg[i] = (const float*)d_in[3 + 4 * i];
        ct[i] = (const float*)d_in[4 + 4 * i];
    }
    const float *lw[3], *lb[3], *lg[3], *lt[3];
    for (int i = 0; i < 3; ++i) {
        lw[i] = (const float*)d_in[33 + 4 * i];
        lb[i] = (const float*)d_in[34 + 4 * i];
        lg[i] = (const float*)d_in[35 + 4 * i];
        lt[i] = (const float*)d_in[36 + 4 * i];
    }

    char* ws = (char*)d_ws;
    float*       Y    = (float*)(ws);                          // 32 MB
    float*       P    = (float*)(ws + (32ull << 20));          // 16 MB
    signed char* X    = (signed char*)(ws + (48ull << 20));    //  8 MB (halo NHWC)
    signed char* WB   = (signed char*)(ws + (56ull << 20));    //  4 MB
    float*       MEAN = (float*)(ws + (60ull << 20));
    float*       INV  = MEAN + 1024;
    float*       Z    = INV + 1024;                            // 256*10 floats

    auto cdiv = [](long a, long b) { return (int)((a + b - 1) / b); };

    const int  CI[8] = {3, 64, 128, 256, 256, 512, 512, 512};
    const int  CO[8] = {64, 128, 256, 256, 512, 512, 512, 512};
    const int  CS[8] = {0, 6, 7, 8, 8, 9, 9, 9};               // log2(Ci)
    const bool PL[8] = {true, true, false, true, false, true, false, true};

    // ---- conv0 (real input) + pool + stats + sign (halo write for conv1) ----
    int H = 16, W = 16;
    {
        const int total = NB * 16 * 16 * 64;
        conv0_pool_kernel<<<cdiv(total, 256), 256, 0, stream>>>(x, cw[0], cb[0], P);
        stats_kernel<<<64, 256, 0, stream>>>(P, MEAN, INV, NB * 16 * 16, 64);
        long padBytes = (long)NB * (H + 2) * (W + 2) * 64;
        fill0_kernel<<<cdiv(padBytes / 16, 256), 256, 0, stream>>>((int4*)X, padBytes / 16);
        bnsign_kernel<<<cdiv((long)total, 256), 256, 0, stream>>>(P, cg[0], ct[0], MEAN, INV,
                                                                  X, NB, H, W, 64, 1);
    }

    // ---- binary conv layers 1..7 (IU8 WMMA implicit-GEMM, async LDS B) ----
    for (int i = 1; i < 8; ++i) {
        const int Ci = CI[i], Co = CO[i], K = 9 * Ci;
        wprep_conv_kernel<<<cdiv((long)Co * K, 256), 256, 0, stream>>>(cw[i], WB, Co, Ci);
        const long Mm = (long)NB * H * W;
        bconv_iu8_kernel<<<(int)((Mm >> 8) * (Co >> 6)), 256, 0, stream>>>(
            X, WB, cb[i], Y, NB, H, W, Ci, Co, CS[i], CS[i] - 6);
        const float* src = Y;
        long Ms = Mm;
        if (PL[i]) {
            pool_kernel<<<cdiv((long)NB * (H / 2) * (W / 2) * Co, 256), 256, 0, stream>>>(
                Y, P, NB, H, W, Co);
            H >>= 1; W >>= 1;
            src = P; Ms = (long)NB * H * W;
        }
        stats_kernel<<<Co, 256, 0, stream>>>(src, MEAN, INV, (int)Ms, Co);
        if (i < 7) {  // next consumer is a conv: halo-padded int8 write
            long padBytes = (long)NB * (H + 2) * (W + 2) * Co;
            fill0_kernel<<<cdiv(padBytes / 16, 256), 256, 0, stream>>>((int4*)X, padBytes / 16);
            bnsign_kernel<<<cdiv(Ms * Co, 256), 256, 0, stream>>>(src, cg[i], ct[i], MEAN, INV,
                                                                  X, NB, H, W, Co, 1);
        } else {      // feeds linear layers: compact [256,512]
            bnsign_kernel<<<cdiv(Ms * Co, 256), 256, 0, stream>>>(src, cg[i], ct[i], MEAN, INV,
                                                                  X, NB, 1, 1, Co, 0);
        }
    }

    // ---- binary linear 0,1 ----
    for (int i = 0; i < 2; ++i) {
        wprep_lin_kernel<<<cdiv(512L * 512, 256), 256, 0, stream>>>(lw[i], WB, 512, 512, 512);
        bgemm_iu8_kernel<<<(256 / 256) * (512 / 64), 256, 0, stream>>>(X, WB, lb[i], Y,
                                                                       256, 512, 512);
        stats_kernel<<<512, 256, 0, stream>>>(Y, MEAN, INV, 256, 512);
        bnsign_kernel<<<cdiv(256L * 512, 256), 256, 0, stream>>>(Y, lg[i], lt[i], MEAN, INV,
                                                                 X, 256, 1, 1, 512, 0);
    }

    // ---- binary linear 2 (512 -> 10, weights zero-padded to 64 rows) ----
    wprep_lin_kernel<<<cdiv(64L * 512, 256), 256, 0, stream>>>(lw[2], WB, 10, 512, 64);
    bgemm_iu8_kernel<<<1, 256, 0, stream>>>(X, WB, lb[2], Y, 256, 10, 512);
    stats_kernel<<<10, 256, 0, stream>>>(Y, MEAN, INV, 256, 10);
    bnapply_kernel<<<cdiv(2560L, 256), 256, 0, stream>>>(Y, lg[2], lt[2], MEAN, INV, Z,
                                                         2560, 10);
    logsoftmax_kernel<<<1, 256, 0, stream>>>(Z, (float*)d_out, 256, 10);
}
